// OrthogonalBivectorBlock_89232240542444
// MI455X (gfx1250) — matrix operand order repaired and checked by reading, not compiled, not run on hardware
//
#include <hip/hip_runtime.h>
#include <hip/hip_bf16.h>
#include <math.h>

typedef __attribute__((ext_vector_type(16))) _Float16 v16h;
typedef __attribute__((ext_vector_type(8)))  _Float16 v8h;
typedef __attribute__((ext_vector_type(8)))  float    v8f;

union V16U { v16h v; v8h h[2]; };

#define B_DIM 2
#define L_DIM 2048
#define D_DIM 512
#define M_DIM (B_DIM * L_DIM)   // 4096 rows total
#define PI_F  3.14159265358979323846f

// ---------------------------------------------------------------------------
// WMMA fragment loaders (16-bit A / B, wave32; contiguous k order, same for A and B)
// A: row-major [M x K] f16. Lane holds row M = lane&15; lanes<16 carry
// K = 0..15 of the 32-chunk, lanes>=16 carry K = 16..31 (contiguous).
__device__ __forceinline__ v16h load_a_frag(const _Float16* A, int ld, int row0,
                                            int k0, int lane) {
    const int row = row0 + (lane & 15);
    const int hi  = lane >> 4;
    const _Float16* p = A + (size_t)row * ld + k0 + 16 * hi;
    return *(const v16h*)(p);
}

// B for Y = X @ W^T with W row-major [N x K]: lane holds column N = lane&15;
// lanes<16 carry K=0..15 of the chunk, lanes>=16 carry K=16..31 (contiguous).
__device__ __forceinline__ v16h load_b_frag(const _Float16* W, int ld, int n0,
                                            int k0, int lane) {
    const int n  = n0 + (lane & 15);
    const int hi = lane >> 4;
    const _Float16* p = W + (size_t)n * ld + k0 + 16 * hi;
    return *(const v16h*)(p);
}

__device__ __forceinline__ v8f wmma_f16(v16h a, v16h b, v8f c) {
    return __builtin_amdgcn_wmma_f32_16x16x32_f16(false, a, false, b,
                                                  (short)0, c, false, false);
}

// ---------------------------------------------------------------------------
__global__ void f32_to_f16_kernel(const float* __restrict__ in,
                                  _Float16* __restrict__ out, int n) {
    int i = blockIdx.x * blockDim.x + threadIdx.x;
    if (i < n) out[i] = (_Float16)in[i];
}

// ---------------------------------------------------------------------------
// WMMA GEMM: C[M,N] = A[M,K] @ W[N,K]^T (+bias[n]) (+residual[M,N]).
// One wave computes a 32x32 macro-tile (2x2 WMMA tiles, 2x fragment reuse);
// 4 waves per block across N. M, N multiples of 32; K multiple of 32.
__global__ void gemm_wmma_kernel(const _Float16* __restrict__ A,
                                 const _Float16* __restrict__ W,
                                 int M, int N, int K,
                                 const float* __restrict__ bias,
                                 const float* __restrict__ residual,
                                 float* __restrict__ Cf,
                                 _Float16* __restrict__ Ch)
{
    const int lane   = threadIdx.x & 31;
    const int wv     = threadIdx.x >> 5;
    const int nMacro = blockIdx.x * 4 + wv;
    if (nMacro * 32 >= N) return;           // wave-uniform: EXEC stays all-1s
    const int n0 = nMacro * 32;
    const int m0 = blockIdx.y * 32;

    v8f c00 = {}, c01 = {}, c10 = {}, c11 = {};
    for (int k0 = 0; k0 < K; k0 += 32) {
        v16h a0 = load_a_frag(A, K, m0,      k0, lane);
        v16h a1 = load_a_frag(A, K, m0 + 16, k0, lane);
        v16h b0 = load_b_frag(W, K, n0,      k0, lane);
        v16h b1 = load_b_frag(W, K, n0 + 16, k0, lane);
        c00 = wmma_f16(a0, b0, c00);
        c01 = wmma_f16(a0, b1, c01);
        c10 = wmma_f16(a1, b0, c10);
        c11 = wmma_f16(a1, b1, c11);
    }

    const int hi = lane >> 4;
    const int nl = lane & 15;
#pragma unroll
    for (int mi = 0; mi < 2; ++mi) {
#pragma unroll
        for (int ni = 0; ni < 2; ++ni) {
            v8f c = (mi == 0) ? (ni == 0 ? c00 : c01) : (ni == 0 ? c10 : c11);
            const int n = n0 + ni * 16 + nl;
#pragma unroll
            for (int r = 0; r < 8; ++r) {
                const int m = m0 + mi * 16 + r + 8 * hi;
                float v = c[r];
                if (bias)     v += bias[n];
                if (residual) v += residual[(size_t)m * N + n];
                if (Cf) Cf[(size_t)m * N + n] = v;
                if (Ch) Ch[(size_t)m * N + n] = (_Float16)v;
            }
        }
    }
}

// ---------------------------------------------------------------------------
// Per-row feature kernel: phasor features, resonance gain, GELU gate; emits
// scaled 64-dim fused features Qf/Kf (f16) for the causal attention.
__global__ void feat_kernel(const float* __restrict__ kqraw,   // [M,32]
                            const float* __restrict__ g1raw,   // [M,256]
                            const float* __restrict__ bk, const float* __restrict__ bq,
                            const float* __restrict__ Wg2, const float* __restrict__ bg2,
                            const float* __restrict__ pos_phases, // [L,16]
                            const float* __restrict__ pos_weight,
                            _Float16* __restrict__ Qf, _Float16* __restrict__ Kf,
                            int L)
{
    const int i   = blockIdx.x;          // row = b*L + l
    const int l   = i % L;
    const int tid = threadIdx.x;         // 64 threads

    __shared__ float s_ck[16], s_sk[16], s_cq[16], s_sq[16], s_res[16];
    __shared__ float s_dot[64];
    __shared__ float s_w1, s_w2;

    if (tid < 16) {
        float kr = kqraw[i * 32 + tid]      + bk[tid];
        float qr = kqraw[i * 32 + 16 + tid] + bq[tid];
        float kp = tanhf(kr) * PI_F;
        float qp = tanhf(qr) * PI_F;
        float ck, sk, cq, sq;
        sincosf(kp, &sk, &ck);
        sincosf(qp, &sq, &cq);
        s_ck[tid] = ck; s_sk[tid] = sk;
        s_cq[tid] = cq; s_sq[tid] = sq;
        s_res[tid] = cosf(kp - qp);
    }
    // exact-erf GELU(g1) . Wg2 partial dot (4 elems per thread)
    float acc = 0.0f;
#pragma unroll
    for (int j = 0; j < 4; ++j) {
        int idx = tid * 4 + j;
        float g = g1raw[i * 256 + idx];
        float ge = 0.5f * g * (1.0f + erff(g * 0.70710678118654752f));
        acc += ge * Wg2[idx];
    }
    s_dot[tid] = acc;
    __syncthreads();

    if (tid == 0) {
        float dot = 0.0f;
        for (int j = 0; j < 64; ++j) dot += s_dot[j];
        float res = 0.0f;
        for (int p = 0; p < 16; ++p) res += s_res[p];
        res *= (1.0f / 16.0f);
        float z    = res + 0.5f;
        float gain = (z > 20.0f) ? z : log1pf(expf(z));
        float gate = 1.0f / (1.0f + expf(-(dot + bg2[0])));
        s_w1 = gate * gain * (1.0f / 16.0f);                       // /P
        s_w2 = (1.0f - gate) / (1.0f + expf(-pos_weight[0]));
    }
    __syncthreads();

    const int g = tid >> 4, p = tid & 15;
    float q, k;
    if (g == 0)      { q = s_w1 * s_cq[p]; k = s_ck[p]; }
    else if (g == 1) { q = s_w1 * s_sq[p]; k = s_sk[p]; }
    else {
        float ph = pos_phases[l * 16 + p];
        float c, s; sincosf(ph, &s, &c);
        if (g == 2) { q = s_w2 * c; k = c; }
        else        { q = s_w2 * s; k = s; }
    }
    Qf[(size_t)i * 64 + tid] = (_Float16)q;
    Kf[(size_t)i * 64 + tid] = (_Float16)k;
}

// ---------------------------------------------------------------------------
__global__ void transpose_v_kernel(const _Float16* __restrict__ V,  // [M,512]
                                   _Float16* __restrict__ VT,       // [B,512,L]
                                   int L) {
    int idx = blockIdx.x * blockDim.x + threadIdx.x;
    if (idx >= M_DIM * D_DIM) return;
    int d   = idx & (D_DIM - 1);
    int row = idx >> 9;
    int b = row / L, l = row % L;
    VT[((size_t)b * D_DIM + d) * L + l] = V[idx];
}

// ---------------------------------------------------------------------------
// Fused causal linear attention: total[l,d] = sum_{t<=l} (Qf[l].Kf[t]) V[t,d].
// 8 waves/block, 256 t per chunk. Phase 1: wave w computes masked score slab
// w (16x32) into LDS. Phase 2: every wave runs AV over all valid slabs for
// its 64 owned D columns, reading A-fragments from LDS.
__global__ void attn_kernel(const _Float16* __restrict__ Qf,   // [M,64]
                            const _Float16* __restrict__ Kf,   // [M,64]
                            const _Float16* __restrict__ VT,   // [B,512,L]
                            float* __restrict__ total,         // [M,512]
                            int L)
{
    const int lane = threadIdx.x & 31;
    const int wv   = threadIdx.x >> 5;          // 0..7
    const int hi   = lane >> 4;
    const int nIdx = lane & 15;

    const int tilesPerBatch = L / 16;
    const int b  = blockIdx.x / tilesPerBatch;
    const int l0 = (blockIdx.x % tilesPerBatch) * 16;
    const int rowBase = b * L;
    const _Float16* vt = VT + (size_t)b * D_DIM * L;   // [512][L]
    const int d0w = wv * 64;

    __shared__ _Float16 ldsA[8][16 * 32];   // 8 score slabs of 16x32

    // Q fragments for this row tile are loop-invariant: hoist (K = 64 = 2 chunks)
    const v16h aQ0 = load_a_frag(Qf, 64, rowBase + l0, 0,  lane);
    const v16h aQ1 = load_a_frag(Qf, 64, rowBase + l0, 32, lane);

    v8f acc[4] = {};
    const int Tend = l0 + 16;               // exclusive causal bound

    for (int T0 = 0; T0 < Tend; T0 += 256) {
        // ---- Phase 1: my score slab (if in range) ----
        const int t0 = T0 + 32 * wv;
        if (t0 < Tend) {
#pragma unroll
            for (int s = 0; s < 2; ++s) {
                const int tb = t0 + 16 * s;
                v8f c = {};
                c = wmma_f16(aQ0, load_b_frag(Kf, 64, rowBase + tb, 0,  lane), c);
                c = wmma_f16(aQ1, load_b_frag(Kf, 64, rowBase + tb, 32, lane), c);
                // causal mask + park as f16 in LDS (C/D layout -> row major)
#pragma unroll
                for (int r = 0; r < 8; ++r) {
                    const int m  = r + 8 * hi;
                    float v = (tb + nIdx <= l0 + m) ? (float)c[r] : 0.0f;
                    ldsA[wv][m * 32 + 16 * s + nIdx] = (_Float16)v;
                }
            }
        }
        __syncthreads();

        // ---- Phase 2: AV over the valid slabs of this chunk ----
        const int nSlab = min(8, (Tend - T0 + 31) >> 5);
        for (int s2 = 0; s2 < nSlab; ++s2) {
            const _Float16* ap = &ldsA[s2][nIdx * 32 + 16 * hi];
            V16U ua;
            ua.h[0] = *(const v8h*)(ap);
            ua.h[1] = *(const v8h*)(ap + 8);
            const int ts = T0 + 32 * s2;
#pragma unroll
            for (int dt = 0; dt < 4; ++dt) {
                v16h bb = load_b_frag(vt, L, d0w + dt * 16, ts, lane);
                acc[dt] = wmma_f16(ua.v, bb, acc[dt]);
            }
        }
        __syncthreads();
    }

#pragma unroll
    for (int dt = 0; dt < 4; ++dt)
#pragma unroll
        for (int r = 0; r < 8; ++r)
            total[(size_t)(rowBase + l0 + r + 8 * hi) * D_DIM + d0w + dt * 16 + nIdx]
                = acc[dt][r];
}

// ---------------------------------------------------------------------------
// Position scale + LayerNorm, emit f16 for the output GEMM.
__global__ void norm_kernel(const float* __restrict__ total,
                            const float* __restrict__ ln_g,
                            const float* __restrict__ ln_b,
                            _Float16* __restrict__ yn, int L)
{
    const int i   = blockIdx.x;
    const int l   = i % L;
    const int tid = threadIdx.x;     // 256
    const float inv = rsqrtf((float)(l + 1) * 4.0f);   // sqrt(pos * planes_per_set)

    float y0 = total[(size_t)i * D_DIM + tid]       * inv;
    float y1 = total[(size_t)i * D_DIM + 256 + tid] * inv;

    __shared__ float ssum[256], ssq[256];
    ssum[tid] = y0 + y1;
    ssq[tid]  = y0 * y0 + y1 * y1;
    __syncthreads();
    for (int o = 128; o > 0; o >>= 1) {
        if (tid < o) { ssum[tid] += ssum[tid + o]; ssq[tid] += ssq[tid + o]; }
        __syncthreads();
    }
    float mean = ssum[0] * (1.0f / 512.0f);
    float var  = ssq[0] * (1.0f / 512.0f) - mean * mean;
    float rstd = rsqrtf(var + 1e-5f);

    yn[(size_t)i * D_DIM + tid]       = (_Float16)((y0 - mean) * rstd * ln_g[tid]       + ln_b[tid]);
    yn[(size_t)i * D_DIM + 256 + tid] = (_Float16)((y1 - mean) * rstd * ln_g[tid + 256] + ln_b[tid + 256]);
}

// ---------------------------------------------------------------------------
static inline size_t alignUp256(size_t x) { return (x + 255) & ~(size_t)255; }

extern "C" void kernel_launch(void* const* d_in, const int* in_sizes, int n_in,
                              void* d_out, int out_size, void* d_ws, size_t ws_size,
                              hipStream_t stream)
{
    (void)in_sizes; (void)n_in; (void)out_size; (void)ws_size;

    const float* x          = (const float*)d_in[0];   // [B,L,512]
    const float* Wk         = (const float*)d_in[1];   // [16,512]
    const float* bk         = (const float*)d_in[2];
    const float* Wq         = (const float*)d_in[3];   // [16,512]
    const float* bq         = (const float*)d_in[4];
    const float* Wv         = (const float*)d_in[5];   // [512,512]
    const float* bv         = (const float*)d_in[6];
    const float* Wg1        = (const float*)d_in[7];   // [256,512]
    const float* bg1        = (const float*)d_in[8];
    const float* Wg2        = (const float*)d_in[9];   // [1,256]
    const float* bg2        = (const float*)d_in[10];
    const float* ln_g       = (const float*)d_in[11];
    const float* ln_b       = (const float*)d_in[12];
    const float* Wo         = (const float*)d_in[13];  // [512,512]
    const float* bo         = (const float*)d_in[14];
    const float* pos_phases = (const float*)d_in[15];  // [2048,16]
    const float* pos_weight = (const float*)d_in[16];

    const int M = M_DIM, L = L_DIM;

    // ---- workspace carve-out ----
    uint8_t* w = (uint8_t*)d_ws;
    _Float16* XH   = (_Float16*)w; w += alignUp256((size_t)M * 512 * 2);
    _Float16* WkqH = (_Float16*)w; w += alignUp256((size_t)32 * 512 * 2);
    _Float16* WvH  = (_Float16*)w; w += alignUp256((size_t)512 * 512 * 2);
    _Float16* Wg1H = (_Float16*)w; w += alignUp256((size_t)256 * 512 * 2);
    _Float16* WoH  = (_Float16*)w; w += alignUp256((size_t)512 * 512 * 2);
    float*    kqraw= (float*)w;    w += alignUp256((size_t)M * 32 * 4);
    float*    g1raw= (float*)w;    w += alignUp256((size_t)M * 256 * 4);
    _Float16* V16  = (_Float16*)w; w += alignUp256((size_t)M * 512 * 2);
    _Float16* VT   = (_Float16*)w; w += alignUp256((size_t)M * 512 * 2);
    _Float16* Qf   = (_Float16*)w; w += alignUp256((size_t)M * 64 * 2);
    _Float16* Kf   = (_Float16*)w; w += alignUp256((size_t)M * 64 * 2);
    float*    totl = (float*)w;    w += alignUp256((size_t)M * 512 * 4);
    _Float16* yn   = (_Float16*)w; w += alignUp256((size_t)M * 512 * 2);

    // ---- f32 -> f16 conversions ----
    auto cvt = [&](const float* src, _Float16* dst, int n) {
        f32_to_f16_kernel<<<(n + 255) / 256, 256, 0, stream>>>(src, dst, n);
    };
    cvt(x,   XH,   M * 512);
    cvt(Wk,  WkqH,           16 * 512);
    cvt(Wq,  WkqH + 16*512,  16 * 512);
    cvt(Wv,  WvH,  512 * 512);
    cvt(Wg1, Wg1H, 256 * 512);
    cvt(Wo,  WoH,  512 * 512);

    const dim3 gblk(128);
    // kq = x @ [Wk;Wq]^T            -> [M,32]
    gemm_wmma_kernel<<<dim3(1, M / 32), gblk, 0, stream>>>(
        XH, WkqH, M, 32, 512, nullptr, nullptr, kqraw, nullptr);
    // V = x @ Wv^T + bv             -> f16 [M,512]
    gemm_wmma_kernel<<<dim3(512 / 128, M / 32), gblk, 0, stream>>>(
        XH, WvH, M, 512, 512, bv, nullptr, nullptr, V16);
    // g1 = x @ Wg1^T + bg1          -> [M,256]
    gemm_wmma_kernel<<<dim3(256 / 128, M / 32), gblk, 0, stream>>>(
        XH, Wg1H, M, 256, 512, bg1, nullptr, g1raw, nullptr);

    feat_kernel<<<M, 64, 0, stream>>>(kqraw, g1raw, bk, bq, Wg2, bg2,
                                      pos_phases, pos_weight, Qf, Kf, L);

    transpose_v_kernel<<<(M * 512 + 255) / 256, 256, 0, stream>>>(V16, VT, L);

    attn_kernel<<<M / 16, 256, 0, stream>>>(Qf, Kf, VT, totl, L);

    norm_kernel<<<M, 256, 0, stream>>>(totl, ln_g, ln_b, yn, L);

    // out = x + yn @ Wo^T + bo
    gemm_wmma_kernel<<<dim3(512 / 128, M / 32), gblk, 0, stream>>>(
        yn, WoH, M, 512, 512, bo, x, (float*)d_out, nullptr);
}
